// SpectralAttention_15650860827415
// MI455X (gfx1250) — compile-verified, matrix operand order checked by reading, not run on hardware
//
#include <hip/hip_runtime.h>

typedef __attribute__((ext_vector_type(16))) _Float16 v16h;
typedef __attribute__((ext_vector_type(8)))  _Float16 v8h;
typedef __attribute__((ext_vector_type(8)))  float    v8f;

#define CIN   256
#define HID   512
#define COUT  256
#define NPIX  130321          // 361*361
#define K1    (2*CIN)         // 512  (interleaved re/im K for layer 1)
#define K2    (2*HID)         // 1024 (interleaved re/im K for layer 2)
#define MT    32              // spectral points per block (2 WMMA p-tiles)
#define XSTR  520             // halves per p-row of X in LDS (512 + 8 pad -> 4-dword bank skew, 16B aligned)
#define HSTR  1032            // halves per p-row of H in LDS (1024 + 8 pad)
#define SMEM_HALVES (MT * XSTR + MT * HSTR)   // 49,664 halves = 99,328 bytes dynamic LDS

union Frag16 { v16h v; v8h h[2]; };
union PackH2 { _Float16 h[2]; unsigned u; };

__device__ __forceinline__ v8f zero8() { return (v8f){0.f,0.f,0.f,0.f,0.f,0.f,0.f,0.f}; }

// ---- weight pre-pack: complex [K][O][2] f32  ->  f16 real-GEMM operands [O][2K] ----
// r-pack row o: [2c]=re(w[c][o]), [2c+1]=-im(w[c][o])   (produces real part)
// i-pack row o: [2c]=im(w[c][o]), [2c+1]= re(w[c][o])   (produces imag part)
__global__ void pack_w_kernel(const float* __restrict__ w, _Float16* __restrict__ wr,
                              _Float16* __restrict__ wi, int K, int O) {
    int idx = blockIdx.x * blockDim.x + threadIdx.x;   // over K*O
    if (idx >= K * O) return;
    int c = idx / O, o = idx % O;
    float re = w[(size_t)idx * 2 + 0];
    float im = w[(size_t)idx * 2 + 1];
    size_t d = (size_t)o * (2 * K) + 2 * c;
    wr[d]     = (_Float16)re;
    wr[d + 1] = (_Float16)(-im);
    wi[d]     = (_Float16)im;
    wi[d + 1] = (_Float16)re;
}

// A fragment (16x32 f16): lane<16 -> row=base+lane,   K in {kk..kk+7, kk+16..kk+23}
//                         lane>=16 -> row=base+lane-16, K in {kk+8..kk+15, kk+24..kk+31}
__device__ __forceinline__ v16h load_a_frag(const _Float16* __restrict__ W, int kstride,
                                            int row, int kk, int hi) {
    Frag16 f;
    const _Float16* base = W + (size_t)row * kstride + kk + hi * 8;
    f.h[0] = *(const v8h*)(base);
    f.h[1] = *(const v8h*)(base + 16);
    return f.v;
}

// B fragment (32x16 f16) from LDS stored as [col p][K']: same K split, col = lane&15
__device__ __forceinline__ v16h load_b_frag(const _Float16* L, int col, int kk, int hi) {
    Frag16 f;
    const _Float16* base = L + col * HSTR + kk + hi * 8;   // dummy; specialized below
    f.h[0] = *(const v8h*)(base);
    f.h[1] = *(const v8h*)(base + 16);
    return f.v;
}

__device__ __forceinline__ v16h load_b_frag_s(const _Float16* L, int stride,
                                              int col, int kk, int hi) {
    Frag16 f;
    const _Float16* base = L + col * stride + kk + hi * 8;
    f.h[0] = *(const v8h*)(base);
    f.h[1] = *(const v8h*)(base + 16);
    return f.v;
}

__device__ __forceinline__ v8f wmma_f16(v16h a, v16h b, v8f c) {
    return __builtin_amdgcn_wmma_f32_16x16x32_f16(false, a, false, b, (short)0, c, false, false);
}

__global__ void __launch_bounds__(256)
spectral_fused_kernel(const float* __restrict__ x,        // [CIN][NPIX][2]
                      const _Float16* __restrict__ w0r,   // [HID][K1]
                      const _Float16* __restrict__ w0i,   // [HID][K1]
                      const _Float16* __restrict__ wqr,   // [COUT][K2]
                      const _Float16* __restrict__ wqi,   // [COUT][K2]
                      float* __restrict__ out) {          // [COUT][NPIX][2]
    extern __shared__ __align__(16) _Float16 smem[];
    _Float16* Xl = smem;                  // [MT][XSTR]  X tile, [p][2c] interleaved re/im
    _Float16* Hl = smem + MT * XSTR;      // [MT][HSTR]  H tile, [p][2h] (relu(re), im)

    const int tid  = threadIdx.x;
    const int lane = tid & 31;
    const int wave = tid >> 5;
    const int lo   = lane & 15;       // N / p column (and A-row offset)
    const int hi   = lane >> 4;       // K-half select
    const int p0   = blockIdx.x * MT;

    // ---- stage X tile: thread t handles p = t&31, channels c = (t>>5) + 8*i ----
    {
        const int p    = tid & 31;
        const int pidx = (p0 + p < NPIX) ? (p0 + p) : (NPIX - 1);  // clamp tail
        for (int c = tid >> 5; c < CIN; c += 8) {
            const float2 v = *(const float2*)(x + ((size_t)c * NPIX + pidx) * 2);
            PackH2 pk;
            pk.h[0] = (_Float16)v.x;
            pk.h[1] = (_Float16)v.y;
            *(unsigned*)&Xl[p * XSTR + 2 * c] = pk.u;
        }
    }
    __syncthreads();

    // ---- phase A: H[512 x 32] = act(W0' * X'); 64 hidden rows x 32 points per wave ----
    {
        v8f accr[4][2], acci[4][2];
#pragma unroll
        for (int t = 0; t < 4; ++t) {
            accr[t][0] = zero8(); accr[t][1] = zero8();
            acci[t][0] = zero8(); acci[t][1] = zero8();
        }
        const int obase = wave * 64;
        for (int kk = 0; kk < K1; kk += 32) {
            // group ALL operand loads of this K-step first (one clause + one wait),
            // then issue 16 back-to-back WMMAs
            const v16h b0 = load_b_frag_s(Xl, XSTR, lo, kk, hi);
            const v16h b1 = load_b_frag_s(Xl + 16 * XSTR, XSTR, lo, kk, hi);
            v16h ar[4], ai[4];
#pragma unroll
            for (int t = 0; t < 4; ++t) {
                const int row = obase + t * 16 + lo;
                ar[t] = load_a_frag(w0r, K1, row, kk, hi);
                ai[t] = load_a_frag(w0i, K1, row, kk, hi);
            }
#pragma unroll
            for (int t = 0; t < 4; ++t) {
                accr[t][0] = wmma_f16(ar[t], b0, accr[t][0]);
                accr[t][1] = wmma_f16(ar[t], b1, accr[t][1]);
                acci[t][0] = wmma_f16(ai[t], b0, acci[t][0]);
                acci[t][1] = wmma_f16(ai[t], b1, acci[t][1]);
            }
        }
        // C layout: VGPR j -> M=j (lanes 0-15) / M=j+8 (lanes 16-31); N = lane&15
#pragma unroll
        for (int t = 0; t < 4; ++t) {
#pragma unroll
            for (int pt = 0; pt < 2; ++pt) {
#pragma unroll
                for (int j = 0; j < 8; ++j) {
                    const int o = obase + t * 16 + j + hi * 8;
                    const int p = pt * 16 + lo;
                    float hr = accr[t][pt][j];
                    hr = hr < 0.f ? 0.f : hr;             // ComplexReLU(mode='real')
                    PackH2 pk;
                    pk.h[0] = (_Float16)hr;
                    pk.h[1] = (_Float16)acci[t][pt][j];
                    *(unsigned*)&Hl[p * HSTR + 2 * o] = pk.u;
                }
            }
        }
    }
    __syncthreads();

    // ---- phase B: OUT[256 x 32] = Wout' * H'; 32 output rows x 32 points per wave ----
    {
        v8f accr[2][2], acci[2][2];
#pragma unroll
        for (int t = 0; t < 2; ++t) {
            accr[t][0] = zero8(); accr[t][1] = zero8();
            acci[t][0] = zero8(); acci[t][1] = zero8();
        }
        const int obase = wave * 32;
        for (int kk = 0; kk < K2; kk += 32) {
            const v16h b0 = load_b_frag_s(Hl, HSTR, lo, kk, hi);
            const v16h b1 = load_b_frag_s(Hl + 16 * HSTR, HSTR, lo, kk, hi);
            v16h ar[2], ai[2];
#pragma unroll
            for (int t = 0; t < 2; ++t) {
                const int row = obase + t * 16 + lo;
                ar[t] = load_a_frag(wqr, K2, row, kk, hi);
                ai[t] = load_a_frag(wqi, K2, row, kk, hi);
            }
#pragma unroll
            for (int t = 0; t < 2; ++t) {
                accr[t][0] = wmma_f16(ar[t], b0, accr[t][0]);
                accr[t][1] = wmma_f16(ar[t], b1, accr[t][1]);
                acci[t][0] = wmma_f16(ai[t], b0, acci[t][0]);
                acci[t][1] = wmma_f16(ai[t], b1, acci[t][1]);
            }
        }
#pragma unroll
        for (int t = 0; t < 2; ++t) {
#pragma unroll
            for (int pt = 0; pt < 2; ++pt) {
#pragma unroll
                for (int j = 0; j < 8; ++j) {
                    const int o = obase + t * 16 + j + hi * 8;
                    const int p = p0 + pt * 16 + lo;
                    if (p < NPIX) {
                        float2 v;
                        v.x = accr[t][pt][j];
                        v.y = acci[t][pt][j];
                        *(float2*)(out + ((size_t)o * NPIX + p) * 2) = v;  // (re, im)
                    }
                }
            }
        }
    }
}

extern "C" void kernel_launch(void* const* d_in, const int* in_sizes, int n_in,
                              void* d_out, int out_size, void* d_ws, size_t ws_size,
                              hipStream_t stream) {
    (void)in_sizes; (void)n_in; (void)out_size; (void)ws_size;
    const float* x    = (const float*)d_in[0];   // [1,256,361,361,2]
    const float* w0   = (const float*)d_in[1];   // [256,512,2]
    const float* wout = (const float*)d_in[2];   // [512,256,2]
    float* out        = (float*)d_out;           // [1,256,361,361,2]

    // workspace layout (f16): w0r | w0i | woutr | wouti  = 2 MB total (L2-resident)
    _Float16* ws  = (_Float16*)d_ws;
    _Float16* w0r = ws;
    _Float16* w0i = ws + (size_t)HID * K1;               // 512*512
    _Float16* wqr = ws + 2 * (size_t)HID * K1;
    _Float16* wqi = ws + 2 * (size_t)HID * K1 + (size_t)COUT * K2;

    // pre-pack weights (deterministic, rebuilt every call)
    {
        int n = CIN * HID;
        pack_w_kernel<<<(n + 255) / 256, 256, 0, stream>>>(w0, w0r, w0i, CIN, HID);
        n = HID * COUT;
        pack_w_kernel<<<(n + 255) / 256, 256, 0, stream>>>(wout, wqr, wqi, HID, COUT);
    }

    // fused two-layer complex spectral MLP: one workgroup per 32 spectral modes
    const int nblocks = (NPIX + MT - 1) / MT;            // 4073
    const size_t smem = (size_t)SMEM_HALVES * sizeof(_Float16);   // 99,328 B dynamic LDS
    spectral_fused_kernel<<<nblocks, 256, smem, stream>>>(x, w0r, w0i, wqr, wqi, out);
}
